// Backbone_26414048871028
// MI455X (gfx1250) — matrix operand order repaired and checked by reading, not compiled, not run on hardware
//
#include <hip/hip_runtime.h>

#define N_NODES 20000
#define E_EDGES 100000
#define CIN     64
#define ATTR    32
#define CH      256
#define FD      1024
#define NBOX    128

typedef __attribute__((ext_vector_type(16))) _Float16 v16h;
typedef __attribute__((ext_vector_type(2)))  _Float16 v2h;
typedef __attribute__((ext_vector_type(8)))  float    v8f;

// ---------------------------------------------------------------- utilities
__global__ void zero_kernel(float* p, long n) {
  long i = (long)blockIdx.x * blockDim.x + threadIdx.x;
  if (i < n) p[i] = 0.0f;
}

// dst[r*ldD + offD + c] = src[r*ldS + offS + c], c in [0,cols)
__global__ void copy_slice_kernel(const float* __restrict__ src, float* __restrict__ dst,
                                  int rows, int ldS, int offS, int ldD, int offD, int cols) {
  long i = (long)blockIdx.x * blockDim.x + threadIdx.x;
  long total = (long)rows * cols;
  if (i < total) {
    int r = (int)(i / cols), c = (int)(i % cols);
    dst[(long)r * ldD + offD + c] = src[(long)r * ldS + offS + c];
  }
}

// W row-major [K,width] f32  ->  Wt [width][K] f16 (transposed)
__global__ void wtrans_kernel(const float* __restrict__ W, _Float16* __restrict__ Wt,
                              int K, int width) {
  long i = (long)blockIdx.x * blockDim.x + threadIdx.x;
  if (i < (long)K * width) {
    int k = (int)(i / width), n = (int)(i % width);
    Wt[(long)n * K + k] = (_Float16)W[i];
  }
}

__global__ void bbox_acc_kernel(const float* __restrict__ featsS, const int* __restrict__ bbox,
                                float* __restrict__ sums, float* __restrict__ counts) {
  long i = (long)blockIdx.x * blockDim.x + threadIdx.x;
  long total = (long)N_NODES * FD;
  if (i < total) {
    int r = (int)(i / FD), c = (int)(i % FD);
    int b = bbox[r];
    atomicAdd(&sums[(long)b * FD + c], featsS[i]);
    if (c == 0) atomicAdd(&counts[b], 1.0f);
  }
}

__global__ void bbox_norm_kernel(float* sums, const float* counts) {
  int i = blockIdx.x * blockDim.x + threadIdx.x;
  if (i < NBOX * FD) sums[i] /= fmaxf(counts[i / FD], 1.0f);
}

// ------------------------------------------------------------- WMMA helpers
// A tile stored in natural order: element K of row r lives at
// As[r*32 + K], so lane (g,rm)'s 16 halves are K = 16g..16g+15 (matches B).
// Identity preserves even-pair adjacency.
__device__ __forceinline__ int aperm(int k) {
  return k;
}

__device__ __forceinline__ v8f wmma_f16(v16h a, v16h b, v8f c) {
  return __builtin_amdgcn_wmma_f32_16x16x32_f16(false, a, false, b, (short)0, c,
                                                false, false);
}

// Async DMA: copy 32B (two b128) of f16 weights straight into LDS.
// ISA: the instruction offset is added to both the LDS and the global address,
// so offset:16 advances both sides.
__device__ __forceinline__ void async_b32B(const _Float16* g, _Float16* lds) {
  unsigned l = (unsigned)(uintptr_t)lds;   // generic->LDS: low 32 bits
  asm volatile("global_load_async_to_lds_b128 %0, %1, off"
               :: "v"(l), "v"(g) : "memory");
  asm volatile("global_load_async_to_lds_b128 %0, %1, off offset:16"
               :: "v"(l), "v"(g) : "memory");
}
__device__ __forceinline__ void wait_async_and_sync() {
  asm volatile("s_wait_asynccnt 0x0" ::: "memory");
  __syncthreads();
}

// B chunk 32(K)x256(N) from transposed f16 weights Wt[width][ldK]:
// thread t -> column bn = t&255, k-range kh = (t>>8)*16; 32B async copy.
__device__ __forceinline__ void load_B_async(const _Float16* __restrict__ Wt, int ldK,
                                             int cb, int k0, int tid, _Float16* Bs) {
  const int bn = tid & 255;
  const int kh = (tid >> 8) * 16;
  async_b32B(Wt + (long)(cb + bn) * ldK + k0 + kh, Bs + bn * 32 + kh);
}

// ----------------------------------------------------- edge message GEMM
// 512 threads = 16 waves; tile = 32 edges x 256 out channels (E%32==0).
// A(edge e) = [fin[dst] (F) | fin[src]-fin[dst] (F) | attr (32)], K = 2F+32.
// m = relu(A@W + bias) * ew[e]; atomic scatter-add into fout[dst].
__global__ __launch_bounds__(512)
void edge_msg_kernel(const float* __restrict__ fin, int ldF, int offF, int F,
                     const float* __restrict__ attrs, const int* __restrict__ edges,
                     const float* __restrict__ ew,
                     const _Float16* __restrict__ Wt, const float* __restrict__ bias,
                     float* __restrict__ fout, int ldO, int offO) {
  __shared__ v16h AsV[32 * 2];     // 32 rows x 32 halves (fragment order)
  __shared__ v16h BsV[256 * 2];    // 256 cols x 32 halves (transposed)
  _Float16* As = (_Float16*)AsV;
  _Float16* Bs = (_Float16*)BsV;

  const int tid = threadIdx.x, lane = tid & 31, wave = tid >> 5;
  const int g = lane >> 4, rm = lane & 15;
  const int ncol = wave * 16 + rm;
  const int e0 = blockIdx.x * 32;
  const int K = 2 * F + ATTR;

  // per-thread A coordinates: one even K-pair of one row (32 rows x 16 pairs)
  const int arow = tid >> 4;            // 0..31
  const int ak   = (tid & 15) * 2;      // even K within chunk
  const int ea = e0 + arow;
  const int sA = edges[ea];
  const int dA = edges[E_EDGES + ea];
  _Float16* ap = As + arow * 32 + aperm(ak);

  v8f acc0 = {}, acc1 = {};
  for (int k0 = 0; k0 < K; k0 += 32) {   // regions are chunk-aligned (F%32==0)
    float v0, v1;
    const int k = k0 + ak;
    if (k0 < F) {
      const float* p = fin + (long)dA * ldF + offF + k;
      v0 = p[0]; v1 = p[1];
    } else if (k0 < 2 * F) {
      const float* ps = fin + (long)sA * ldF + offF + (k - F);
      const float* pd = fin + (long)dA * ldF + offF + (k - F);
      v0 = ps[0] - pd[0]; v1 = ps[1] - pd[1];
    } else {
      const float* p = attrs + (long)ea * ATTR + (k - 2 * F);
      v0 = p[0]; v1 = p[1];
    }
    v2h t; t[0] = (_Float16)v0; t[1] = (_Float16)v1;
    *(v2h*)ap = t;
    load_B_async(Wt, K, 0, k0, tid, Bs);
    wait_async_and_sync();
    {
      v16h b  = *(const v16h*)(Bs + ncol * 32 + g * 16);
      v16h a0 = *(const v16h*)(As + rm * 32 + g * 16);
      v16h a1 = *(const v16h*)(As + (16 + rm) * 32 + g * 16);
      acc0 = wmma_f16(a0, b, acc0);
      acc1 = wmma_f16(a1, b, acc1);
    }
    __syncthreads();
  }

  const float bv = bias[ncol];
#pragma unroll
  for (int r = 0; r < 8; ++r) {          // C/D layout: M = r + 8g
    int e = e0 + r + 8 * g;
    float v = fmaxf(acc0[r] + bv, 0.0f) * ew[e];
    atomicAdd(&fout[(long)edges[E_EDGES + e] * ldO + offO + ncol], v);
    int e2 = e + 16;
    float w = fmaxf(acc1[r] + bv, 0.0f) * ew[e2];
    atomicAdd(&fout[(long)edges[E_EDGES + e2] * ldO + offO + ncol], w);
  }
}

// ----------------------------------------------------- dense node GEMM
// out[row, cb+n] = relu([src1|src2](row) @ W[:, cb+n] + bias[cb+n])
// grid.x = row tiles of 32 (rows%32==0), grid.y = 256-wide column groups.
__global__ __launch_bounds__(512)
void node_gemm_kernel(const float* __restrict__ src1, int ld1, int off1, int C1,
                      const float* __restrict__ src2, int ld2, int off2,
                      const _Float16* __restrict__ Wt,
                      const float* __restrict__ bias,
                      float* __restrict__ out, int ldO, int offO, int K) {
  __shared__ v16h AsV[32 * 2];
  __shared__ v16h BsV[256 * 2];
  _Float16* As = (_Float16*)AsV;
  _Float16* Bs = (_Float16*)BsV;

  const int tid = threadIdx.x, lane = tid & 31, wave = tid >> 5;
  const int g = lane >> 4, rm = lane & 15;
  const int r0 = blockIdx.x * 32;
  const int cb = blockIdx.y * 256;
  const int ncol = wave * 16 + rm;
  const int arow = tid >> 4;
  const int ak   = (tid & 15) * 2;
  const int ar = r0 + arow;
  _Float16* ap = As + arow * 32 + aperm(ak);

  v8f acc0 = {}, acc1 = {};
  for (int k0 = 0; k0 < K; k0 += 32) {   // C1 is chunk-aligned
    const int k = k0 + ak;
    const float* p = (k0 < C1) ? src1 + (long)ar * ld1 + off1 + k
                               : src2 + (long)ar * ld2 + off2 + (k - C1);
    v2h t; t[0] = (_Float16)p[0]; t[1] = (_Float16)p[1];
    *(v2h*)ap = t;
    load_B_async(Wt, K, cb, k0, tid, Bs);
    wait_async_and_sync();
    {
      v16h b  = *(const v16h*)(Bs + ncol * 32 + g * 16);
      v16h a0 = *(const v16h*)(As + rm * 32 + g * 16);
      v16h a1 = *(const v16h*)(As + (16 + rm) * 32 + g * 16);
      acc0 = wmma_f16(a0, b, acc0);
      acc1 = wmma_f16(a1, b, acc1);
    }
    __syncthreads();
  }

  const float bv = bias[cb + ncol];
#pragma unroll
  for (int r = 0; r < 8; ++r) {
    int row = r0 + r + 8 * g;
    out[(long)row * ldO + offO + cb + ncol]        = fmaxf(acc0[r] + bv, 0.0f);
    out[(long)(row + 16) * ldO + offO + cb + ncol] = fmaxf(acc1[r] + bv, 0.0f);
  }
}

// ---------------------------------------------------------------- launcher
extern "C" void kernel_launch(void* const* d_in, const int* in_sizes, int n_in,
                              void* d_out, int out_size, void* d_ws, size_t ws_size,
                              hipStream_t stream) {
  (void)in_sizes; (void)n_in; (void)out_size; (void)ws_size;
  const float* x      = (const float*)d_in[0];
  const int*   edges  = (const int*)  d_in[1];
  const float* ew     = (const float*)d_in[2];
  const float* attrs  = (const float*)d_in[3];
  const int*   bbox   = (const int*)  d_in[4];
  // d_in[5] = num_boxes (==128, compile-time constant here)
  const float* W_msg0 = (const float*)d_in[6];
  const float* b_msg0 = (const float*)d_in[7];
  const float* W_sup0 = (const float*)d_in[8];
  const float* b_sup0 = (const float*)d_in[9];
  const float* W_msg  = (const float*)d_in[10];   // [3][544][256]
  const float* b_msg  = (const float*)d_in[11];   // [3][256]
  const float* W_sup  = (const float*)d_in[12];   // [3][512][256]
  const float* b_sup  = (const float*)d_in[13];   // [3][256]
  const float* W_fuse = (const float*)d_in[14];   // [1024][1024]
  const float* b_fuse = (const float*)d_in[15];
  const float* W_fsup = (const float*)d_in[16];
  const float* b_fsup = (const float*)d_in[17];
  float* out = (float*)d_out;

  // workspace layout
  float* feats  = (float*)d_ws;                         // [N,1024] == feats_cat
  float* featsS = feats  + (long)N_NODES * FD;          // [N,1024] == fs_cat
  float* sums   = featsS + (long)N_NODES * FD;          // [128,1024] -> fs_mean
  float* counts = sums   + (long)NBOX * FD;             // [128]
  _Float16* Wt_msg0 = (_Float16*)(counts + NBOX);       // [256][160]
  _Float16* Wt_sup0 = Wt_msg0 + 256 * 160;              // [256][320]
  _Float16* Wt_msg  = Wt_sup0 + 256 * 320;              // 3 x [256][544]
  _Float16* Wt_sup  = Wt_msg  + 3 * 256 * 544;          // 3 x [256][512]
  _Float16* Wt_fuse = Wt_sup  + 3 * 256 * 512;          // [1024][1024]
  _Float16* Wt_fsup = Wt_fuse + 1024 * 1024;            // [1024][1024]

  const int edgeTiles = E_EDGES / 32;                   // 3125 (exact)
  const int rowTiles  = N_NODES / 32;                   // 625  (exact)

  // 0) transpose + f16-convert all weights (reused by thousands of tiles)
  {
    auto WT = [&](const float* W, _Float16* Wt, int K, int width) {
      long n = (long)K * width;
      wtrans_kernel<<<(unsigned)((n + 511) / 512), 512, 0, stream>>>(W, Wt, K, width);
    };
    WT(W_msg0, Wt_msg0, 160, CH);
    WT(W_sup0, Wt_sup0, 320, CH);
    for (int i = 0; i < 3; ++i) {
      WT(W_msg + (long)i * 544 * CH, Wt_msg + (long)i * CH * 544, 544, CH);
      WT(W_sup + (long)i * 512 * CH, Wt_sup + (long)i * CH * 512, 512, CH);
    }
    WT(W_fuse, Wt_fuse, FD, FD);
    WT(W_fsup, Wt_fsup, FD, FD);
  }

  // 1) head GraphConv into feats slice 0
  {
    long n = (long)N_NODES * FD;                        // zero whole feats buffer
    zero_kernel<<<(unsigned)((n + 511) / 512), 512, 0, stream>>>(feats, n);
  }
  edge_msg_kernel<<<edgeTiles, 512, 0, stream>>>(
      x, CIN, 0, CIN, attrs, edges, ew, Wt_msg0, b_msg0, feats, FD, 0);

  // 2) head super node: relu([f0 | x] @ W_sup0)
  node_gemm_kernel<<<dim3(rowTiles, 1), 512, 0, stream>>>(
      feats, FD, 0, CH, x, CIN, 0, Wt_sup0, b_sup0,
      featsS, FD, 0, CH + CIN);

  // 3) ResBlocks
  for (int i = 1; i < 4; ++i) {
    // residual init: feats slice i = feats slice i-1
    {
      long n = (long)N_NODES * CH;
      copy_slice_kernel<<<(unsigned)((n + 511) / 512), 512, 0, stream>>>(
          feats, feats, N_NODES, FD, (i - 1) * CH, FD, i * CH, CH);
    }
    edge_msg_kernel<<<edgeTiles, 512, 0, stream>>>(
        feats, FD, (i - 1) * CH, CH, attrs, edges, ew,
        Wt_msg + (long)(i - 1) * CH * 544, b_msg + (i - 1) * CH,
        feats, FD, i * CH);
    node_gemm_kernel<<<dim3(rowTiles, 1), 512, 0, stream>>>(
        feats, FD, i * CH, CH, featsS, FD, (i - 1) * CH,
        Wt_sup + (long)(i - 1) * CH * 512, b_sup + (i - 1) * CH,
        featsS, FD, i * CH, 2 * CH);
  }

  // 4) fusion: out[:, 0:1024] = relu(feats_cat @ W_fuse + b)
  node_gemm_kernel<<<dim3(rowTiles, 4), 512, 0, stream>>>(
      feats, FD, 0, FD, feats, FD, 0, Wt_fuse, b_fuse,
      out, 2 * FD, 0, FD);
  // out[:, 1024:2048] = feats_cat
  {
    long n = (long)N_NODES * FD;
    copy_slice_kernel<<<(unsigned)((n + 511) / 512), 512, 0, stream>>>(
        feats, out, N_NODES, FD, 0, 2 * FD, FD, FD);
  }

  // 5) box branch: segment mean of featsS over bbox_idx
  {
    long n = (long)NBOX * FD + NBOX;
    zero_kernel<<<(unsigned)((n + 511) / 512), 512, 0, stream>>>(sums, n);
  }
  {
    long n = (long)N_NODES * FD;
    bbox_acc_kernel<<<(unsigned)((n + 511) / 512), 512, 0, stream>>>(
        featsS, bbox, sums, counts);
  }
  bbox_norm_kernel<<<(NBOX * FD + 511) / 512, 512, 0, stream>>>(sums, counts);

  float* outS = out + (long)N_NODES * 2 * FD;
  node_gemm_kernel<<<dim3(NBOX / 32, 4), 512, 0, stream>>>(
      sums, FD, 0, FD, sums, FD, 0, Wt_fsup, b_fsup,
      outS, 2 * FD, 0, FD);
  {
    long n = (long)NBOX * FD;
    copy_slice_kernel<<<(unsigned)((n + 511) / 512), 512, 0, stream>>>(
        sums, outS, NBOX, FD, 0, 2 * FD, FD, FD);
  }
}
